// Seq2Seq_74156905333342
// MI455X (gfx1250) — matrix operand order repaired and checked by reading, not compiled, not run on hardware
//
#include <hip/hip_runtime.h>
#include <hip/hip_bf16.h>

#define N_VOCAB   50257
#define N_CLASSES 50257
#define E_DIM     256
#define H_DIM     512
#define S_IN      512
#define S_DEC     512      // h1 + 511 decode steps
#define START_TOK 1

#define FC_BM     128                     // M rows per workgroup (LDS panel)
#define FC_BN     256                     // N cols per workgroup
#define FC_NB     ((N_CLASSES + FC_BN - 1) / FC_BN)   // 197
// fp32 A panel in LDS with TDM padding: +4 dwords every 256 dwords
// -> effective pitch 520 dwords, pads only at the k=256 boundary.
#define A_PITCH32 520
#define FC_LDS_BYTES (FC_BM * A_PITCH32 * 4)          // 266,240 B (< 320 KB)

typedef __attribute__((ext_vector_type(16))) __bf16 v16bf;
typedef __attribute__((ext_vector_type(8)))  float  v8f;
typedef unsigned int u32x4 __attribute__((ext_vector_type(4)));
typedef int          i32x8 __attribute__((ext_vector_type(8)));
typedef int          i32x4 __attribute__((ext_vector_type(4)));

__device__ __forceinline__ int pad_adj(int col) { return (col >= 256) ? 4 : 0; }

// ---------------------------------------------------------------------------
// A fragment (16x32) from the padded fp32 LDS panel, cvt to bf16 in-register.
// lane half h, element e -> k = kk + 8h + e (e<8), kk + 8h + 8 + e (e>=8)
__device__ __forceinline__ v16bf frag_a_lds32(const float* As, int row, int kk,
                                              int half) {
    int k0 = kk + half * 8;      // 8-aligned, run never crosses k=256
    int k1 = k0 + 16;
    const float* p0 = As + row * A_PITCH32 + k0 + pad_adj(k0);
    const float* p1 = As + row * A_PITCH32 + k1 + pad_adj(k1);
    float4 x0 = *(const float4*)(p0);
    float4 x1 = *(const float4*)(p0 + 4);
    float4 x2 = *(const float4*)(p1);
    float4 x3 = *(const float4*)(p1 + 4);
    v16bf r;
    r[0]=(__bf16)x0.x; r[1]=(__bf16)x0.y; r[2]=(__bf16)x0.z; r[3]=(__bf16)x0.w;
    r[4]=(__bf16)x1.x; r[5]=(__bf16)x1.y; r[6]=(__bf16)x1.z; r[7]=(__bf16)x1.w;
    r[8]=(__bf16)x2.x; r[9]=(__bf16)x2.y; r[10]=(__bf16)x2.z; r[11]=(__bf16)x2.w;
    r[12]=(__bf16)x3.x; r[13]=(__bf16)x3.y; r[14]=(__bf16)x3.z; r[15]=(__bf16)x3.w;
    return r;
}

// A fragment straight from row-major fp32 global (used by xw_gemm).
__device__ __forceinline__ v16bf frag_a(const float* __restrict__ A, int lda,
                                        int m, int kk, int half) {
    const float* p = A + (size_t)m * lda + kk + half * 8;
    float4 x0 = *(const float4*)(p);
    float4 x1 = *(const float4*)(p + 4);
    float4 x2 = *(const float4*)(p + 16);
    float4 x3 = *(const float4*)(p + 20);
    v16bf r;
    r[0]=(__bf16)x0.x; r[1]=(__bf16)x0.y; r[2]=(__bf16)x0.z; r[3]=(__bf16)x0.w;
    r[4]=(__bf16)x1.x; r[5]=(__bf16)x1.y; r[6]=(__bf16)x1.z; r[7]=(__bf16)x1.w;
    r[8]=(__bf16)x2.x; r[9]=(__bf16)x2.y; r[10]=(__bf16)x2.z; r[11]=(__bf16)x2.w;
    r[12]=(__bf16)x3.x; r[13]=(__bf16)x3.y; r[14]=(__bf16)x3.z; r[15]=(__bf16)x3.w;
    return r;
}

// B fragment (32x16) from row-major weight W[N][K] (B[k][n] = W[n][k]).
// lane half h, element e -> k = kk + 8h + e (e<8), kk + 8h + 8 + e (e>=8)
__device__ __forceinline__ v16bf frag_b(const float* __restrict__ W, int ldk,
                                        int n, int kk, int half, bool valid) {
    v16bf r;
    if (valid) {
        const float* p = W + (size_t)n * ldk + kk + half * 8;
        float4 x0 = *(const float4*)(p);
        float4 x1 = *(const float4*)(p + 4);
        float4 x2 = *(const float4*)(p + 16);
        float4 x3 = *(const float4*)(p + 20);
        r[0]=(__bf16)x0.x; r[1]=(__bf16)x0.y; r[2]=(__bf16)x0.z; r[3]=(__bf16)x0.w;
        r[4]=(__bf16)x1.x; r[5]=(__bf16)x1.y; r[6]=(__bf16)x1.z; r[7]=(__bf16)x1.w;
        r[8]=(__bf16)x2.x; r[9]=(__bf16)x2.y; r[10]=(__bf16)x2.z; r[11]=(__bf16)x2.w;
        r[12]=(__bf16)x3.x; r[13]=(__bf16)x3.y; r[14]=(__bf16)x3.z; r[15]=(__bf16)x3.w;
    } else {
#pragma unroll
        for (int i = 0; i < 16; ++i) r[i] = (__bf16)0.0f;
    }
    return r;
}

// ---------------------------------------------------------------------------
// 1) Embedding gather: Xenc[512,256], Xdec[512,256] (row0 = START embedding)
__global__ void gather_embed(const int* __restrict__ in_ids,
                             const int* __restrict__ out_ids,
                             const float* __restrict__ emb,
                             float* __restrict__ Xenc,
                             float* __restrict__ Xdec) {
    int row = blockIdx.x;       // 0..1023
    int col = threadIdx.x;      // 0..255
    if (row < S_IN) {
        int id = in_ids[row];
        Xenc[(size_t)row * E_DIM + col] = emb[(size_t)id * E_DIM + col];
    } else {
        int r  = row - S_IN;
        int id = (r == 0) ? START_TOK : out_ids[r - 1];
        Xdec[(size_t)r * E_DIM + col] = emb[(size_t)id * E_DIM + col];
    }
}

// ---------------------------------------------------------------------------
// 2) Transpose W_hh (512x512) so the serial RNN reads are coalesced.
__global__ void transpose_whh(const float* __restrict__ we,
                              const float* __restrict__ wd,
                              float* __restrict__ te,
                              float* __restrict__ td) {
    __shared__ float tile[16][17];
    const float* in  = blockIdx.z ? wd : we;
    float*       out = blockIdx.z ? td : te;
    int bx = blockIdx.x * 16, by = blockIdx.y * 16;
    tile[threadIdx.y][threadIdx.x] =
        in[(size_t)(by + threadIdx.y) * H_DIM + bx + threadIdx.x];
    __syncthreads();
    out[(size_t)(bx + threadIdx.y) * H_DIM + by + threadIdx.x] =
        tile[threadIdx.x][threadIdx.y];
}

// ---------------------------------------------------------------------------
// 3) Batched input projection: XW = X @ W_ih^T + b_ih   (M=512,N=512,K=256)
__global__ void __launch_bounds__(128)
xw_gemm(const float* __restrict__ Xe, const float* __restrict__ Xd,
        const float* __restrict__ wih_e, const float* __restrict__ wih_d,
        const float* __restrict__ bih_e, const float* __restrict__ bih_d,
        float* __restrict__ XWe, float* __restrict__ XWd) {
    const int wave = threadIdx.x >> 5;
    const int lane = threadIdx.x & 31;
    const int half = lane >> 4;
    const int l16  = lane & 15;

    int tile = blockIdx.x * 4 + wave;       // 0..1023 (32x32 tiles)
    int tM = tile & 31, tN = tile >> 5;

    const float* X   = blockIdx.y ? Xd : Xe;
    const float* Wih = blockIdx.y ? wih_d : wih_e;
    const float* Bih = blockIdx.y ? bih_d : bih_e;
    float*       XW  = blockIdx.y ? XWd : XWe;

    int n = tN * 16 + l16;
    int m = tM * 16 + l16;

    v8f c;
    float bv = Bih[n];
#pragma unroll
    for (int r = 0; r < 8; ++r) c[r] = bv;

#pragma unroll
    for (int kk = 0; kk < E_DIM; kk += 32) {
        v16bf a = frag_a(X, E_DIM, m, kk, half);
        v16bf b = frag_b(Wih, E_DIM, n, kk, half, true);
        c = __builtin_amdgcn_wmma_f32_16x16x32_bf16(
                false, a, false, b, (short)0, c, false, false);
    }
#pragma unroll
    for (int r = 0; r < 8; ++r)
        XW[(size_t)(tM * 16 + half * 8 + r) * H_DIM + n] = c[r];
}

// ---------------------------------------------------------------------------
// 4) Serial recurrence: latency-bound, one persistent workgroup, h in LDS.
__global__ void __launch_bounds__(H_DIM)
rnn_seq(const float* __restrict__ h0,
        const float* __restrict__ XWe, const float* __restrict__ XWd,
        const float* __restrict__ whhT_e, const float* __restrict__ whhT_d,
        const float* __restrict__ bhh_e, const float* __restrict__ bhh_d,
        float* __restrict__ pred) {
    __shared__ float h[H_DIM];
    const int t = threadIdx.x;
    h[t] = h0[t];
    __syncthreads();
    const float be = bhh_e[t];
    const float bd = bhh_d[t];

    for (int s = 0; s < S_IN; ++s) {
        float acc = XWe[(size_t)s * H_DIM + t] + be;
#pragma unroll 8
        for (int k = 0; k < H_DIM; ++k)
            acc = fmaf(whhT_e[(size_t)k * H_DIM + t], h[k], acc);
        __syncthreads();
        h[t] = tanhf(acc);
        __syncthreads();
    }
    for (int s = 0; s < S_DEC; ++s) {
        float acc = XWd[(size_t)s * H_DIM + t] + bd;
#pragma unroll 8
        for (int k = 0; k < H_DIM; ++k)
            acc = fmaf(whhT_d[(size_t)k * H_DIM + t], h[k], acc);
        __syncthreads();
        float hv = tanhf(acc);
        h[t] = hv;
        pred[(size_t)s * H_DIM + t] = hv;
        __syncthreads();
    }
}

// ---------------------------------------------------------------------------
// 5) FC head: out = pred @ fc_w^T + fc_b   (M=512, N=50257, K=512).
//    A panel staged into LDS by the Tensor Data Mover (tensor_load_to_lds,
//    TENSORcnt), with TDM LDS-padding for bank spread. 8 waves, block tile
//    128x256, wave tile 64x64 (16 accumulators), 16 v_wmma per K-step.
__global__ void __launch_bounds__(256)
fc_gemm(const float* __restrict__ predg, const float* __restrict__ fcw,
        const float* __restrict__ fcb, float* __restrict__ out) {
    extern __shared__ __align__(16) unsigned char smem_raw[];
    float* As = (float*)smem_raw;       // [FC_BM][A_PITCH32] fp32 (padded)

    const int tid  = threadIdx.x;
    const int lane = tid & 31;
    const int wave = tid >> 5;          // 0..7
    const int half = lane >> 4;
    const int l16  = lane & 15;

    const int blockN = blockIdx.x;      // 0..FC_NB-1
    const int blockM = blockIdx.y;      // 0..3

    // ---- Stage A panel (FC_BM x 512 fp32) into LDS via the TDM ----
#if __has_builtin(__builtin_amdgcn_tensor_load_to_lds) && \
    __has_builtin(__builtin_amdgcn_s_wait_tensorcnt)
    if (wave == 0) {
        unsigned long long gaddr =
            (unsigned long long)(const void*)(predg + (size_t)blockM * FC_BM * H_DIM);
        unsigned lds_addr = (unsigned)(unsigned long long)(void*)As; // LDS offset
        // D# group0: count=1 | lds_addr | global_addr[56:0] | type=2
        u32x4 g0;
        g0[0] = 1u;
        g0[1] = lds_addr;
        g0[2] = (unsigned)(gaddr & 0xFFFFFFFFull);
        g0[3] = ((unsigned)(gaddr >> 32) & 0x1FFFFFFu) | (2u << 30);
        // D# group1: data_size=4B, pad_enable, pad_interval=256dw, pad_amount=4dw,
        // tensor 512x512 (stride0=512), tile 512x128.
        i32x8 g1;
        g1[0] = (2 << 16) | (1 << 20) | (7 << 22) | (3 << 25);
        g1[1] = (H_DIM & 0xFFFF) << 16;              // tensor_dim0 lo
        g1[2] = (H_DIM & 0xFFFF) << 16;              // tensor_dim0 hi=0 | dim1 lo
        g1[3] = (H_DIM & 0xFFFF) << 16;              // dim1 hi=0 | tile_dim0=512
        g1[4] = FC_BM;                               // tile_dim1=128, tile_dim2=0
        g1[5] = H_DIM;                               // tensor_dim0_stride lo
        g1[6] = 0;
        g1[7] = 0;
        i32x4 gz4 = {0, 0, 0, 0};                    // groups 2/3 unused (2D)
        i32x8 gz8 = {0, 0, 0, 0, 0, 0, 0, 0};        // trailing group (clang-23 ABI)
        __builtin_amdgcn_tensor_load_to_lds(g0, g1, gz4, gz4, gz8, 0);
        __builtin_amdgcn_s_wait_tensorcnt(0);
    }
#else
    {
        const float* Ag = predg + (size_t)blockM * FC_BM * H_DIM;
        for (int i = tid * 4; i < FC_BM * H_DIM; i += 256 * 4) {
            int row = i >> 9;
            int col = i & (H_DIM - 1);
            float4 v = *(const float4*)(Ag + (size_t)row * H_DIM + col);
            *(float4*)(As + row * A_PITCH32 + col + pad_adj(col)) = v;
        }
    }
#endif
    __syncthreads();

    // ---- Wave sub-tile: 64(M) x 64(N) ----
    const int wm = wave & 1;            // 2 M-groups of 64
    const int wn = wave >> 1;           // 4 N-groups of 64
    const int n0 = blockN * FC_BN + wn * 64;

    int  ncol[4];
    bool nv[4];
    v8f  acc[4][4];                     // [mi][ni] : 128 VGPRs
#pragma unroll
    for (int ni = 0; ni < 4; ++ni) {
        int n = n0 + ni * 16 + l16;
        ncol[ni] = n;
        nv[ni]   = (n < N_CLASSES);
        float bv = nv[ni] ? fcb[n] : 0.0f;
#pragma unroll
        for (int mi = 0; mi < 4; ++mi)
#pragma unroll
            for (int r = 0; r < 8; ++r) acc[mi][ni][r] = bv;
    }

    for (int kk = 0; kk < H_DIM; kk += 32) {
        v16bf a0 = frag_a_lds32(As, wm * 64 + 0 * 16 + l16, kk, half);
        v16bf a1 = frag_a_lds32(As, wm * 64 + 1 * 16 + l16, kk, half);
        v16bf a2 = frag_a_lds32(As, wm * 64 + 2 * 16 + l16, kk, half);
        v16bf a3 = frag_a_lds32(As, wm * 64 + 3 * 16 + l16, kk, half);
#pragma unroll
        for (int ni = 0; ni < 4; ++ni) {
            v16bf b = frag_b(fcw, H_DIM, ncol[ni], kk, half, nv[ni]);
            acc[0][ni] = __builtin_amdgcn_wmma_f32_16x16x32_bf16(
                false, a0, false, b, (short)0, acc[0][ni], false, false);
            acc[1][ni] = __builtin_amdgcn_wmma_f32_16x16x32_bf16(
                false, a1, false, b, (short)0, acc[1][ni], false, false);
            acc[2][ni] = __builtin_amdgcn_wmma_f32_16x16x32_bf16(
                false, a2, false, b, (short)0, acc[2][ni], false, false);
            acc[3][ni] = __builtin_amdgcn_wmma_f32_16x16x32_bf16(
                false, a3, false, b, (short)0, acc[3][ni], false, false);
        }
    }

    const int gm0 = blockM * FC_BM + wm * 64;
#pragma unroll
    for (int mi = 0; mi < 4; ++mi)
#pragma unroll
        for (int ni = 0; ni < 4; ++ni)
            if (nv[ni])
#pragma unroll
                for (int r = 0; r < 8; ++r)
                    out[(size_t)(gm0 + mi * 16 + half * 8 + r) * N_CLASSES +
                        ncol[ni]] = acc[mi][ni][r];
}

// ---------------------------------------------------------------------------
extern "C" void kernel_launch(void* const* d_in, const int* in_sizes, int n_in,
                              void* d_out, int out_size, void* d_ws, size_t ws_size,
                              hipStream_t stream) {
    const float* h0         = (const float*)d_in[0];
    const int*   input_ids  = (const int*)d_in[1];
    const int*   output_ids = (const int*)d_in[2];
    const float* emb        = (const float*)d_in[3];
    const float* enc_wih    = (const float*)d_in[4];
    const float* enc_whh    = (const float*)d_in[5];
    const float* enc_bih    = (const float*)d_in[6];
    const float* enc_bhh    = (const float*)d_in[7];
    const float* dec_wih    = (const float*)d_in[8];
    const float* dec_whh    = (const float*)d_in[9];
    const float* dec_bih    = (const float*)d_in[10];
    const float* dec_bhh    = (const float*)d_in[11];
    const float* fc_w       = (const float*)d_in[12];
    const float* fc_b       = (const float*)d_in[13];
    float* out = (float*)d_out;

    // Workspace layout (floats). pred aliases the X region (X dead after xw_gemm).
    float* ws     = (float*)d_ws;
    float* Xenc   = ws;                 // 512*256
    float* Xdec   = ws + 131072;        // 512*256
    float* pred   = ws;                 // 512*512 (alias over Xenc|Xdec)
    float* XWe    = ws + 262144;        // 512*512
    float* XWd    = ws + 524288;        // 512*512
    float* whhT_e = ws + 786432;        // 512*512
    float* whhT_d = ws + 1048576;       // 512*512  (total ~5 MB)

    gather_embed<<<S_IN + S_DEC, E_DIM, 0, stream>>>(
        input_ids, output_ids, emb, Xenc, Xdec);

    transpose_whh<<<dim3(H_DIM / 16, H_DIM / 16, 2), dim3(16, 16), 0, stream>>>(
        enc_whh, dec_whh, whhT_e, whhT_d);

    xw_gemm<<<dim3(256, 2), 128, 0, stream>>>(
        Xenc, Xdec, enc_wih, dec_wih, enc_bih, dec_bih, XWe, XWd);

    rnn_seq<<<1, H_DIM, 0, stream>>>(
        h0, XWe, XWd, whhT_e, whhT_d, enc_bhh, dec_bhh, pred);

    fc_gemm<<<dim3(FC_NB, 4), 256, FC_LDS_BYTES, stream>>>(pred, fc_w, fc_b, out);
}
